// SSMBlock_24240795419333
// MI455X (gfx1250) — matrix operand order repaired and checked by reading, not compiled, not run on hardware
//
#include <hip/hip_runtime.h>
#include <hip/hip_bf16.h>
#include <math.h>

#define DMODEL 1024
#define DINNER 2048
#define BSZ    2
#define SEQ    2048
#define ROWS   (BSZ * SEQ)   // 4096
#define NSTATE 16
#define KCONV  4

typedef __attribute__((ext_vector_type(16))) __bf16 v16bf;
typedef __attribute__((ext_vector_type(8)))  float  v8f;

// ---------- helpers ----------
__device__ __forceinline__ unsigned short f2bf(float f) {
  unsigned int u = __float_as_uint(f);
  u += 0x7fffu + ((u >> 16) & 1u);   // round-to-nearest-even
  return (unsigned short)(u >> 16);
}
__device__ __forceinline__ float silu_(float x) {
  return x * (1.0f / (1.0f + __expf(-x)));
}

struct __align__(16) U4 { unsigned int a, b, c, d; };
__device__ __forceinline__ U4 ld16(const unsigned short* p) { return *(const U4*)p; }

struct Frag { union { v16bf v; U4 q[2]; }; };

// A fragment (16x32 bf16): lanes 0-15 hold K=k0..k0+7 & k0+16..k0+23; lanes 16-31 +8.
// K is always a multiple of 32 -> unconditional 16B loads.
__device__ __forceinline__ Frag load_a(const unsigned short* row, int k0, int half) {
  Frag f;
  int c0 = k0 + half * 8;
  f.q[0] = ld16(row + c0);
  f.q[1] = ld16(row + c0 + 16);
  return f;
}
// B fragment (32x16 bf16, stored as Bt rows = columns of B): same K order as A (k0+8*half, +16).
__device__ __forceinline__ Frag load_b(const unsigned short* row, int k0, int half) {
  Frag f;
  int c0 = k0 + half * 8;
  f.q[0] = ld16(row + c0);
  f.q[1] = ld16(row + c0 + 16);
  return f;
}

// ---------- epilogues ----------
struct EpiStore {
  float* C; int ldc;
  __device__ void operator()(int m, int n, float v) const { C[(size_t)m * ldc + n] = v; }
};
struct EpiScaleRow {        // Bx = (xs @ B^T) * delta[row]
  float* C; const float* s; int ldc;
  __device__ void operator()(int m, int n, float v) const { C[(size_t)m * ldc + n] = v * s[m]; }
};
struct EpiY {               // y = hs@C^T + D*xs ; y *= silu(gate) ; -> bf16
  const float* xsf; const float* Dv; const float* xg; unsigned short* yb;
  __device__ void operator()(int m, int n, float v) const {
    float t = v + Dv[n] * xsf[(size_t)m * DINNER + n];
    float g = xg[(size_t)m * (2 * DINNER) + DINNER + n];
    t *= silu_(g);
    yb[(size_t)m * DINNER + n] = f2bf(t);
  }
};
struct EpiRes {             // out = residual + y@W_out
  const float* res; float* out;
  __device__ void operator()(int m, int n, float v) const {
    out[(size_t)m * DMODEL + n] = res[(size_t)m * DMODEL + n] + v;
  }
};

// ---------- WMMA GEMM: C[M,N] = A[M,K](bf16,row) x Bt[N,K](bf16,row) ----------
// 256 threads = 8 waves; macro-tile (128*MT) x (16*NT).
// Per 32-wide k-step: issue all MT+NT fragment loads, then MT*NT WMMAs, so the
// scheduler can overlap loads with the XDL burst via partial s_wait_loadcnt.
template <int K, int MT, int NT, typename Epi>
__global__ __launch_bounds__(256) void wmma_gemm(
    const unsigned short* __restrict__ A, const unsigned short* __restrict__ Bt,
    int lda, int ldb, Epi epi) {
  const int lane = threadIdx.x & 31;
  const int wave = threadIdx.x >> 5;
  const int half = lane >> 4;
  const int lid  = lane & 15;
  const int m0 = blockIdx.y * (128 * MT) + wave * (16 * MT);
  const int n0 = blockIdx.x * (16 * NT);

  const unsigned short* arow[MT];
#pragma unroll
  for (int i = 0; i < MT; ++i) arow[i] = A + (size_t)(m0 + i * 16 + lid) * lda;
  const unsigned short* brow[NT];
#pragma unroll
  for (int j = 0; j < NT; ++j) brow[j] = Bt + (size_t)(n0 + j * 16 + lid) * ldb;

  const v8f vz = {0.f, 0.f, 0.f, 0.f, 0.f, 0.f, 0.f, 0.f};
  v8f acc[MT][NT];
#pragma unroll
  for (int i = 0; i < MT; ++i)
#pragma unroll
    for (int j = 0; j < NT; ++j) acc[i][j] = vz;

#pragma unroll 2
  for (int k0 = 0; k0 < K; k0 += 32) {
    // Speculative prefetch: OOB translation failures are silently dropped (ISA TH),
    // so no bounds guard -> no per-iteration branch.
#pragma unroll
    for (int i = 0; i < MT; ++i)
      __builtin_prefetch(arow[i] + k0 + 64, 0, 1);   // global_prefetch_b8

    Frag a[MT];
#pragma unroll
    for (int i = 0; i < MT; ++i) a[i] = load_a(arow[i], k0, half);
    Frag b[NT];
#pragma unroll
    for (int j = 0; j < NT; ++j) b[j] = load_b(brow[j], k0, half);

#pragma unroll
    for (int j = 0; j < NT; ++j)
#pragma unroll
      for (int i = 0; i < MT; ++i)
        acc[i][j] = __builtin_amdgcn_wmma_f32_16x16x32_bf16(
            false, a[i].v, false, b[j].v, (short)0, acc[i][j], false, false);
  }
  // C/D layout: VGPR r -> (M = m0 + 16*i + r + 8*half, N = n0 + 16*j + lid)
#pragma unroll
  for (int i = 0; i < MT; ++i)
#pragma unroll
    for (int j = 0; j < NT; ++j)
#pragma unroll
      for (int r = 0; r < 8; ++r)
        epi(m0 + i * 16 + r + half * 8, n0 + j * 16 + lid, acc[i][j][r]);
}

// ---------- LayerNorm -> bf16 ----------
__global__ __launch_bounds__(256) void k_layernorm(
    const float* __restrict__ x, const float* __restrict__ g,
    const float* __restrict__ b, unsigned short* __restrict__ out) {
  int row = blockIdx.x;
  const float* xr = x + (size_t)row * DMODEL;
  float s = 0.f, s2 = 0.f;
  for (int i = threadIdx.x; i < DMODEL; i += 256) { float v = xr[i]; s += v; s2 += v * v; }
  __shared__ float sh[512];
  sh[threadIdx.x] = s; sh[256 + threadIdx.x] = s2;
  __syncthreads();
  for (int o = 128; o > 0; o >>= 1) {
    if (threadIdx.x < o) { sh[threadIdx.x] += sh[threadIdx.x + o]; sh[256 + threadIdx.x] += sh[256 + threadIdx.x + o]; }
    __syncthreads();
  }
  float mu  = sh[0] / DMODEL;
  float var = sh[256] / DMODEL - mu * mu;
  float inv = rsqrtf(var + 1e-5f);
  for (int i = threadIdx.x; i < DMODEL; i += 256)
    out[(size_t)row * DMODEL + i] = f2bf((xr[i] - mu) * inv * g[i] + b[i]);
}

// ---------- f32(KxN) -> bf16 transpose (NxK) ----------
__global__ __launch_bounds__(256) void k_transpose_bf16(
    const float* __restrict__ in, unsigned short* __restrict__ out, int K, int N) {
  __shared__ float tile[32][33];
  int kb = blockIdx.x * 32, nb = blockIdx.y * 32;
  int tx = threadIdx.x & 31, ty = threadIdx.x >> 5;
  for (int r = ty; r < 32; r += 8) {
    int k = kb + r, n = nb + tx;
    tile[r][tx] = (k < K && n < N) ? in[(size_t)k * N + n] : 0.f;
  }
  __syncthreads();
  for (int r = ty; r < 32; r += 8) {
    int n = nb + r, k = kb + tx;
    if (n < N && k < K) out[(size_t)n * K + k] = f2bf(tile[tx][r]);
  }
}

// ---------- plain f32 -> bf16 ----------
__global__ void k_f32_to_bf16(const float* __restrict__ in, unsigned short* __restrict__ out, int n) {
  int i = blockIdx.x * blockDim.x + threadIdx.x;
  if (i < n) out[i] = f2bf(in[i]);
}

// ---------- f32(rows x kin) -> bf16(rows x kout), zero-padded K ----------
__global__ void k_pad_bf16(const float* __restrict__ in, unsigned short* __restrict__ out,
                           int rows, int kin, int kout) {
  int i = blockIdx.x * blockDim.x + threadIdx.x;
  if (i >= rows * kout) return;
  int r = i / kout, k = i - r * kout;
  out[i] = (k < kin) ? f2bf(in[(size_t)r * kin + k]) : (unsigned short)0;
}

// ---------- causal depthwise conv (K=4) + SiLU ----------
__global__ __launch_bounds__(256) void k_conv_silu(
    const float* __restrict__ xg, const float* __restrict__ cw, const float* __restrict__ cb,
    float* __restrict__ xsf, unsigned short* __restrict__ xsb) {
  int idx = blockIdx.x * blockDim.x + threadIdx.x;
  if (idx >= ROWS * DINNER) return;
  int d  = idx & (DINNER - 1);
  int bl = idx >> 11;            // / DINNER
  int l  = bl & (SEQ - 1);
  int b  = bl >> 11;             // / SEQ
  float acc = cb[d];
#pragma unroll
  for (int j = 0; j < KCONV; ++j) {
    int lt = l - (KCONV - 1) + j;
    if (lt >= 0) acc += cw[d * KCONV + j] * xg[((size_t)(b * SEQ + lt)) * (2 * DINNER) + d];
  }
  float s = silu_(acc);
  xsf[idx] = s;
  xsb[idx] = f2bf(s);
}

// ---------- delta = mean(softplus(dpre + bd)) per row ----------
__global__ __launch_bounds__(256) void k_delta(
    const float* __restrict__ dpre, const float* __restrict__ bd, float* __restrict__ delta) {
  int row = blockIdx.x;
  const float* r = dpre + (size_t)row * DINNER;
  float s = 0.f;
  for (int i = threadIdx.x; i < DINNER; i += 256) {
    float v = r[i] + bd[i];
    s += (v > 20.f) ? v : log1pf(__expf(v));
  }
  __shared__ float sh[256];
  sh[threadIdx.x] = s;
  __syncthreads();
  for (int o = 128; o > 0; o >>= 1) {
    if (threadIdx.x < o) sh[threadIdx.x] += sh[threadIdx.x + o];
    __syncthreads();
  }
  if (threadIdx.x == 0) delta[row] = sh[0] / DINNER;
}

// ---------- dA = exp(delta[row] * (-exp(A_log[n]))) ----------
__global__ void k_dA(const float* __restrict__ delta, const float* __restrict__ A_log,
                     float* __restrict__ dA, int n) {
  int i = blockIdx.x * blockDim.x + threadIdx.x;
  if (i >= n) return;
  int st = i & (NSTATE - 1);
  int row = i >> 4;
  dA[i] = __expf(delta[row] * (-__expf(A_log[st])));
}

// ---------- sequential scan: 32 independent (batch,state) recurrences ----------
__global__ void k_scan(const float* __restrict__ dA, const float* __restrict__ Bx,
                       float* __restrict__ hs) {
  int t = threadIdx.x;            // 0..31
  int b = t >> 4, n = t & 15;
  float h = 0.f;
  for (int l = 0; l < SEQ; ++l) {
    size_t idx = ((size_t)(b * SEQ + l)) * NSTATE + n;
    h = dA[idx] * h + Bx[idx];
    hs[idx] = h;
  }
}

// ---------- host launcher ----------
extern "C" void kernel_launch(void* const* d_in, const int* in_sizes, int n_in,
                              void* d_out, int out_size, void* d_ws, size_t ws_size,
                              hipStream_t stream) {
  const float* x      = (const float*)d_in[0];
  const float* gln    = (const float*)d_in[1];
  const float* bln    = (const float*)d_in[2];
  const float* W_in   = (const float*)d_in[3];
  const float* conv_w = (const float*)d_in[4];
  const float* conv_b = (const float*)d_in[5];
  const float* A_log  = (const float*)d_in[6];
  const float* B_mat  = (const float*)d_in[7];
  const float* C_mat  = (const float*)d_in[8];
  const float* D_vec  = (const float*)d_in[9];
  const float* Wd     = (const float*)d_in[10];
  const float* bd     = (const float*)d_in[11];
  const float* W_out  = (const float*)d_in[12];
  float* out = (float*)d_out;

  char* p = (char*)d_ws;
  auto alloc = [&](size_t bytes) { void* r = (void*)p; p += (bytes + 255) & ~(size_t)255; return r; };
  unsigned short* h_bf  = (unsigned short*)alloc((size_t)ROWS * DMODEL * 2);
  unsigned short* WinT  = (unsigned short*)alloc((size_t)(2 * DINNER) * DMODEL * 2); // 4096x1024
  unsigned short* WdT   = (unsigned short*)alloc((size_t)DINNER * DINNER * 2);       // 2048x2048
  unsigned short* WoutT = (unsigned short*)alloc((size_t)DMODEL * DINNER * 2);       // 1024x2048
  unsigned short* Bm_bf = (unsigned short*)alloc((size_t)NSTATE * DINNER * 2);       // 16x2048 (N x K)
  unsigned short* CmP   = (unsigned short*)alloc((size_t)DINNER * 32 * 2);           // 2048x32 (K padded)
  float*          xg    = (float*)alloc((size_t)ROWS * 2 * DINNER * 4);
  float*          xsf   = (float*)alloc((size_t)ROWS * DINNER * 4);
  unsigned short* xsb   = (unsigned short*)alloc((size_t)ROWS * DINNER * 2);
  float*          dpre  = (float*)alloc((size_t)ROWS * DINNER * 4);
  float*          delta = (float*)alloc((size_t)ROWS * 4);
  float*          dAb   = (float*)alloc((size_t)ROWS * NSTATE * 4);
  float*          Bx    = (float*)alloc((size_t)ROWS * NSTATE * 4);
  float*          hs    = (float*)alloc((size_t)ROWS * NSTATE * 4);
  unsigned short* hsP   = (unsigned short*)alloc((size_t)ROWS * 32 * 2);             // 4096x32 (K padded)
  unsigned short* yb    = (unsigned short*)alloc((size_t)ROWS * DINNER * 2);

  // 1) LayerNorm -> bf16 activations
  k_layernorm<<<ROWS, 256, 0, stream>>>(x, gln, bln, h_bf);

  // 2) weight conversions (N x K bf16)
  k_transpose_bf16<<<dim3(DMODEL / 32, (2 * DINNER) / 32), 256, 0, stream>>>(W_in, WinT, DMODEL, 2 * DINNER);
  k_transpose_bf16<<<dim3(DINNER / 32, DINNER / 32), 256, 0, stream>>>(Wd, WdT, DINNER, DINNER);
  k_transpose_bf16<<<dim3(DINNER / 32, DMODEL / 32), 256, 0, stream>>>(W_out, WoutT, DINNER, DMODEL);
  k_f32_to_bf16<<<(NSTATE * DINNER + 255) / 256, 256, 0, stream>>>(B_mat, Bm_bf, NSTATE * DINNER);
  k_pad_bf16<<<(DINNER * 32 + 255) / 256, 256, 0, stream>>>(C_mat, CmP, DINNER, NSTATE, 32);

  // 3) xg = h @ W_in      [4096 x 4096], K=1024
  wmma_gemm<DMODEL, 2, 4, EpiStore><<<dim3((2 * DINNER) / 64, ROWS / 256), 256, 0, stream>>>(
      h_bf, WinT, DMODEL, DMODEL, EpiStore{xg, 2 * DINNER});

  // 4) depthwise causal conv + SiLU -> xsf(f32), xsb(bf16)
  k_conv_silu<<<(ROWS * DINNER) / 256, 256, 0, stream>>>(xg, conv_w, conv_b, xsf, xsb);

  // 5) dpre = xs @ Wd     [4096 x 2048], K=2048
  wmma_gemm<DINNER, 2, 4, EpiStore><<<dim3(DINNER / 64, ROWS / 256), 256, 0, stream>>>(
      xsb, WdT, DINNER, DINNER, EpiStore{dpre, DINNER});

  // 6) delta = mean(softplus(dpre + bd))
  k_delta<<<ROWS, 256, 0, stream>>>(dpre, bd, delta);

  // 7) dA
  k_dA<<<(ROWS * NSTATE + 255) / 256, 256, 0, stream>>>(delta, A_log, dAb, ROWS * NSTATE);

  // 8) Bx = (xs @ B^T) * delta[row]   [4096 x 16], K=2048
  wmma_gemm<DINNER, 2, 1, EpiScaleRow><<<dim3(1, ROWS / 256), 256, 0, stream>>>(
      xsb, Bm_bf, DINNER, DINNER, EpiScaleRow{Bx, delta, NSTATE});

  // 9) scan (32 independent recurrences), then pad-convert hs to K=32 bf16
  k_scan<<<1, 32, 0, stream>>>(dAb, Bx, hs);
  k_pad_bf16<<<(ROWS * 32 + 255) / 256, 256, 0, stream>>>(hs, hsP, ROWS, NSTATE, 32);

  // 10) y = (hs @ C^T + D*xs) * silu(gate)  [4096 x 2048], K=32 (zero-padded) -> bf16
  wmma_gemm<32, 2, 4, EpiY><<<dim3(DINNER / 64, ROWS / 256), 256, 0, stream>>>(
      hsP, CmP, 32, 32, EpiY{xsf, D_vec, xg, yb});

  // 11) out = residual + y @ W_out   [4096 x 1024], K=2048
  wmma_gemm<DINNER, 2, 4, EpiRes><<<dim3(DMODEL / 64, ROWS / 256), 256, 0, stream>>>(
      yb, WoutT, DINNER, DINNER, EpiRes{x, out});
}
